// VectorQuantization_61907658605313
// MI455X (gfx1250) — matrix operand order repaired and checked by reading, not compiled, not run on hardware
//
#include <hip/hip_runtime.h>
#include <hip/hip_bf16.h>

// ---------------------------------------------------------------------------
// VQ-VAE vector quantization for MI455X (gfx1250, wave32, WMMA).
// Distances via D = e2[n] - 2 * (X @ C^T) using v_wmma_f32_16x16x32_bf16.
// ---------------------------------------------------------------------------

typedef __attribute__((ext_vector_type(16))) __bf16 v16bf;
typedef __attribute__((ext_vector_type(8)))  float  v8f;

union AU { v16bf v; __bf16 e[16]; uint4 q[2]; };
union CU { v8f   v; float  e[8]; };

#define N_ROWS   65536      // 64*32*32
#define K_CODES  512
#define DIMS     64
#define CB_PAD   72         // padded row stride (halves) -> 144B, kills LDS bank conflicts
#define FLT_BIG  3.402823466e+38f

// ---- kernel 0: zero accumulators (must run every call; buffers not re-poisoned)
__global__ void __launch_bounds__(512) vq_zero_kernel(float* __restrict__ sse,
                                                      float* __restrict__ counts) {
    counts[threadIdx.x] = 0.0f;
    if (threadIdx.x == 0) *sse = 0.0f;
}

// ---- kernel 1: codebook fp32 -> bf16 (padded) + exact fp32 |e|^2
__global__ void __launch_bounds__(64) vq_prep_kernel(const float* __restrict__ cb,
                                                     unsigned short* __restrict__ cb16p,
                                                     float* __restrict__ e2) {
    const int row = blockIdx.x;      // 0..511
    const int t   = threadIdx.x;     // 0..63
    const float v = cb[row * DIMS + t];
    const __bf16 b = (__bf16)v;
    cb16p[row * CB_PAD + t] = __builtin_bit_cast(unsigned short, b);

    float s = v * v;
#pragma unroll
    for (int m = 1; m < 32; m <<= 1) s += __shfl_xor(s, m, 32);
    __shared__ float sh[2];
    if ((t & 31) == 0) sh[t >> 5] = s;
    __syncthreads();
    if (t == 0) e2[row] = sh[0] + sh[1];
}

// ---- kernel 2: WMMA distance + argmin.
// Block = 256 threads = 8 waves; each wave owns one 16-row M-tile.
// Grid = 65536/(16*8) = 512 blocks.
__global__ void __launch_bounds__(256) vq_argmin_kernel(
    const float* __restrict__ x,
    const unsigned short* __restrict__ cb16p,
    const float* __restrict__ e2,
    int* __restrict__ idx_out)
{
    __shared__ uint4 lcbq[(K_CODES * CB_PAD * 2) / 16];   // 73728 B
    __shared__ float se2[K_CODES];                        // 2048 B

    {   // cooperative stage of padded bf16 codebook + |e|^2 into LDS
        const uint4* src = (const uint4*)cb16p;
        for (int i = threadIdx.x; i < (K_CODES * CB_PAD * 2) / 16; i += 256) lcbq[i] = src[i];
        for (int i = threadIdx.x; i < K_CODES; i += 256) se2[i] = e2[i];
    }
    __syncthreads();

    const unsigned short* lcb = (const unsigned short*)lcbq;
    const int lane = threadIdx.x & 31;
    const int wv   = threadIdx.x >> 5;
    const int hlf  = lane >> 4;      // which 16-lane half
    const int sub  = lane & 15;      // M row (A) / N column (B)
    const size_t rowBase = ((size_t)blockIdx.x * 8 + wv) * 16;

    // A fragments (16x32 bf16, two K-blocks covering D=64), per ISA layout:
    // lanes 0-15: M=lane, K = {0..7, 16..23}; lanes 16-31: M=lane-16, K = {8..15, 24..31}
    AU a0, a1;
    {
        const float* xr = x + (rowBase + sub) * DIMS;
        const int k0 = hlf * 8;
        float4 p0 = *(const float4*)(xr + k0);
        float4 p1 = *(const float4*)(xr + k0 + 4);
        float4 p2 = *(const float4*)(xr + k0 + 16);
        float4 p3 = *(const float4*)(xr + k0 + 20);
        float4 p4 = *(const float4*)(xr + k0 + 32);
        float4 p5 = *(const float4*)(xr + k0 + 36);
        float4 p6 = *(const float4*)(xr + k0 + 48);
        float4 p7 = *(const float4*)(xr + k0 + 52);
        const float f0[16] = {p0.x,p0.y,p0.z,p0.w, p1.x,p1.y,p1.z,p1.w,
                              p2.x,p2.y,p2.z,p2.w, p3.x,p3.y,p3.z,p3.w};
        const float f1[16] = {p4.x,p4.y,p4.z,p4.w, p5.x,p5.y,p5.z,p5.w,
                              p6.x,p6.y,p6.z,p6.w, p7.x,p7.y,p7.z,p7.w};
#pragma unroll
        for (int j = 0; j < 16; ++j) { a0.e[j] = (__bf16)f0[j]; a1.e[j] = (__bf16)f1[j]; }
    }

    float bestV[8]; int bestI[8];
#pragma unroll
    for (int i = 0; i < 8; ++i) { bestV[i] = FLT_BIG; bestI[i] = 0; }

    for (int t = 0; t < K_CODES / 16; ++t) {          // 32 code tiles
        const int code = t * 16 + sub;
        // B fragment (32x16 bf16): lane = column N; lanes 0-15 hold K {0..7, 16..23},
        // lanes 16-31 hold K {8..15, 24..31} (same order as A).
        const uint4* bp0 = (const uint4*)(lcb + code * CB_PAD + hlf * 8);
        const uint4* bp1 = (const uint4*)(lcb + code * CB_PAD + 32 + hlf * 8);
        AU b0, b1;
        b0.q[0] = bp0[0]; b0.q[1] = bp0[2];
        b1.q[0] = bp1[0]; b1.q[1] = bp1[2];

        CU c;
#pragma unroll
        for (int i = 0; i < 8; ++i) c.e[i] = 0.0f;
        c.v = __builtin_amdgcn_wmma_f32_16x16x32_bf16(false, a0.v, false, b0.v,
                                                      (short)0, c.v, false, false);
        c.v = __builtin_amdgcn_wmma_f32_16x16x32_bf16(false, a1.v, false, b1.v,
                                                      (short)0, c.v, false, false);

        const float en = se2[code];
#pragma unroll
        for (int i = 0; i < 8; ++i) {           // C layout: vgpr i -> M = i (+8 for hi half)
            float s = en - 2.0f * c.e[i];
            if (s < bestV[i]) { bestV[i] = s; bestI[i] = code; }
        }
    }

    // min+index reduction across the 16 lanes of each half (columns), first-index ties
#pragma unroll
    for (int m = 1; m < 16; m <<= 1) {
#pragma unroll
        for (int i = 0; i < 8; ++i) {
            float ov = __shfl_xor(bestV[i], m, 32);
            int   oi = __shfl_xor(bestI[i], m, 32);
            if (ov < bestV[i] || (ov == bestV[i] && oi < bestI[i])) {
                bestV[i] = ov; bestI[i] = oi;
            }
        }
    }
    if (sub == 0) {                 // lane 0 -> rows 0..7, lane 16 -> rows 8..15
#pragma unroll
        for (int i = 0; i < 8; ++i)
            idx_out[rowBase + hlf * 8 + i] = bestI[i];
    }
}

// ---- kernel 3: gather quantized (fp32, exact), fused SSE + histogram.
// Block = 256 threads handles 16 rows; grid = 4096.
__global__ void __launch_bounds__(256) vq_gather_kernel(
    const float* __restrict__ x, const float* __restrict__ cb,
    const int* __restrict__ idx, float* __restrict__ out,
    float* __restrict__ sse, float* __restrict__ counts)
{
    const int t = threadIdx.x;
    const size_t row = (size_t)blockIdx.x * 16 + (t >> 4);
    const int coff = (t & 15) * 4;
    const int k = idx[row];

    float4 q  = *(const float4*)(cb + (size_t)k * DIMS + coff);
    float4 xv = *(const float4*)(x + row * DIMS + coff);
    *(float4*)(out + row * DIMS + coff) = q;   // == straight-through forward value

    float dx = q.x - xv.x, dy = q.y - xv.y, dz = q.z - xv.z, dw = q.w - xv.w;
    float s = dx*dx + dy*dy + dz*dz + dw*dw;
#pragma unroll
    for (int m = 1; m < 32; m <<= 1) s += __shfl_xor(s, m, 32);
    __shared__ float sh[8];
    if ((t & 31) == 0) sh[t >> 5] = s;
    __syncthreads();
    if (t == 0) {
        float tot = 0.0f;
#pragma unroll
        for (int i = 0; i < 8; ++i) tot += sh[i];
        atomicAdd(sse, tot);
    }
    if ((t & 15) == 0) atomicAdd(&counts[k], 1.0f);  // one per row
}

// ---- kernel 4: finalize loss + perplexity
__global__ void __launch_bounds__(512) vq_final_kernel(
    const float* __restrict__ sse, const float* __restrict__ counts,
    float* __restrict__ out_tail)
{
    const int t = threadIdx.x;          // 0..511 == code id
    float p = counts[t] * (1.0f / (float)N_ROWS);
    float s = p * logf(p + 1e-10f);
#pragma unroll
    for (int m = 1; m < 32; m <<= 1) s += __shfl_xor(s, m, 32);
    __shared__ float sh[16];
    if ((t & 31) == 0) sh[t >> 5] = s;
    __syncthreads();
    if (t == 0) {
        float tot = 0.0f;
#pragma unroll
        for (int i = 0; i < 16; ++i) tot += sh[i];
        // q_latent_loss == e_latent_loss numerically -> loss = 1.25 * MSE
        out_tail[0] = 1.25f * sse[0] * (1.0f / (float)(N_ROWS * DIMS));
        out_tail[1] = expf(-tot);
    }
}

extern "C" void kernel_launch(void* const* d_in, const int* in_sizes, int n_in,
                              void* d_out, int out_size, void* d_ws, size_t ws_size,
                              hipStream_t stream) {
    (void)in_sizes; (void)n_in; (void)out_size; (void)ws_size;
    const float* x  = (const float*)d_in[0];   // inputs   [64,32,32,64] fp32
    const float* cb = (const float*)d_in[1];   // codebook [512,64]      fp32
    float* out = (float*)d_out;                // [4194304 quantized | loss | perplexity]

    char* ws = (char*)d_ws;
    unsigned short* cb16p = (unsigned short*)ws;              // 512*72*2 = 73728 B
    float* e2     = (float*)(ws + 73728);                     // 2048 B
    int*   idx    = (int*)  (ws + 75776);                     // 262144 B
    float* counts = (float*)(ws + 337920);                    // 2048 B
    float* sse    = (float*)(ws + 339968);                    // 4 B

    vq_zero_kernel  <<<1,    512, 0, stream>>>(sse, counts);
    vq_prep_kernel  <<<512,   64, 0, stream>>>(cb, cb16p, e2);
    vq_argmin_kernel<<<512,  256, 0, stream>>>(x, cb16p, e2, idx);
    vq_gather_kernel<<<4096, 256, 0, stream>>>(x, cb, idx, out, sse, counts);
    vq_final_kernel <<<1,    512, 0, stream>>>(sse, counts, out + (size_t)N_ROWS * DIMS);
}
